// Decoder_7808250544206
// MI455X (gfx1250) — matrix operand order repaired and checked by reading, not compiled, not run on hardware
//
#include <hip/hip_runtime.h>
#include <hip/hip_bf16.h>

typedef __attribute__((ext_vector_type(16))) __bf16 v16bf;
typedef __attribute__((ext_vector_type(8)))  float  v8f;

#define BATCH 64
#define SEQ   2048
#define HDIM  512
#define EDIM  256
#define VOC   32000
#define NSPLIT 8   // S-dim splits for attention partial pass

// ---------------------------------------------------------------------------
// Attention pass 1: online-softmax partials over a chunk of S (reads hse ONCE)
// grid = B * NSPLIT workgroups, 256 threads (8 waves)
// hse stream is NON-TEMPORAL: 256 MB > 192 MB L2; keep weights resident instead.
// ---------------------------------------------------------------------------
__global__ void __launch_bounds__(256)
attn_partial_kernel(const float* __restrict__ hsd,   // (B,H)
                    const float* __restrict__ hse,   // (B,S,H)
                    float* __restrict__ pm,          // (B*NSPLIT)
                    float* __restrict__ pl,          // (B*NSPLIT)
                    float* __restrict__ pacc)        // (B*NSPLIT, H)
{
  __shared__ float q[HDIM];
  __shared__ float wacc[8][HDIM];
  __shared__ float wm[8], wl[8];

  const int b     = blockIdx.x / NSPLIT;
  const int chunk = blockIdx.x % NSPLIT;
  const int tid   = threadIdx.x;
  const int lane  = tid & 31;
  const int wave  = tid >> 5;
  const int srows = SEQ / NSPLIT;           // 256 rows per WG
  const int s0    = chunk * srows;

  for (int d = tid; d < HDIM; d += 256) q[d] = hsd[b * HDIM + d];
  __syncthreads();

  float qreg[16];
#pragma unroll
  for (int k = 0; k < 16; ++k) qreg[k] = q[lane + 32 * k];

  float m = -INFINITY, l = 0.f;
  float acc[16];
#pragma unroll
  for (int k = 0; k < 16; ++k) acc[k] = 0.f;

  const float* base = hse + (size_t)b * SEQ * HDIM;
  for (int s = s0 + wave; s < s0 + srows; s += 8) {
    const float* row = base + (size_t)s * HDIM;
    // cover the full next 2KB row across the 32 lanes (64B per lane)
    __builtin_prefetch(row + 8 * HDIM + lane * 16, 0, 0);
    float rv[16];
    float dot = 0.f;
#pragma unroll
    for (int k = 0; k < 16; ++k) {              // coalesced: lane strides d
      rv[k] = __builtin_nontemporal_load(row + lane + 32 * k);
      dot = fmaf(qreg[k], rv[k], dot);
    }
    // wave32 all-reduce of dot
    for (int off = 16; off >= 1; off >>= 1) dot += __shfl_xor(dot, off, 32);
    // online softmax update
    float mn = fmaxf(m, dot);
    float sc = __expf(m - mn);
    float p  = __expf(dot - mn);
    l = l * sc + p;
#pragma unroll
    for (int k = 0; k < 16; ++k) acc[k] = fmaf(acc[k], sc, p * rv[k]);
    m = mn;
  }

#pragma unroll
  for (int k = 0; k < 16; ++k) wacc[wave][lane + 32 * k] = acc[k];
  if (lane == 0) { wm[wave] = m; wl[wave] = l; }
  __syncthreads();

  // merge 8 waves -> one partial per WG
  float M = -INFINITY;
#pragma unroll
  for (int w = 0; w < 8; ++w) M = fmaxf(M, wm[w]);
  float L = 0.f;
#pragma unroll
  for (int w = 0; w < 8; ++w) L += wl[w] * __expf(wm[w] - M);

  const int pidx = b * NSPLIT + chunk;
  if (tid == 0) { pm[pidx] = M; pl[pidx] = L; }
  for (int d = tid; d < HDIM; d += 256) {
    float s = 0.f;
#pragma unroll
    for (int w = 0; w < 8; ++w) s = fmaf(wacc[w][d], __expf(wm[w] - M), s);
    pacc[(size_t)pidx * HDIM + d] = s;   // un-normalized weighted sum
  }
}

// ---------------------------------------------------------------------------
// Attention pass 2: merge partials -> ctx, plus embedding gather
// writes inp = concat([emb[x], ctx]) as (B, E+H)
// ---------------------------------------------------------------------------
__global__ void __launch_bounds__(256)
attn_finalize_kernel(const int* __restrict__ x,
                     const float* __restrict__ emb,
                     const float* __restrict__ pm,
                     const float* __restrict__ pl,
                     const float* __restrict__ pacc,
                     float* __restrict__ inp)
{
  const int b = blockIdx.x;
  const int tid = threadIdx.x;
  float M = -INFINITY;
#pragma unroll
  for (int w = 0; w < NSPLIT; ++w) M = fmaxf(M, pm[b * NSPLIT + w]);
  float L = 0.f;
#pragma unroll
  for (int w = 0; w < NSPLIT; ++w) L += pl[b * NSPLIT + w] * __expf(pm[b * NSPLIT + w] - M);
  const float invL = 1.f / L;

  for (int d = tid; d < HDIM; d += 256) {
    float s = 0.f;
#pragma unroll
    for (int w = 0; w < NSPLIT; ++w)
      s = fmaf(pacc[(size_t)(b * NSPLIT + w) * HDIM + d], __expf(pm[b * NSPLIT + w] - M), s);
    inp[b * (EDIM + HDIM) + EDIM + d] = s * invL;
  }
  const int tok = x[b];
  for (int e = tid; e < EDIM; e += 256)
    inp[b * (EDIM + HDIM) + e] = emb[(size_t)tok * EDIM + e];
}

// ---------------------------------------------------------------------------
// WMMA GEMM: Out(M=64, N) = A0(64,K0) @ W0(N,K0)^T [+ A1(64,K1) @ W1(N,K1)^T] + bias
// f32 memory, bf16 WMMA inputs, f32 accumulate.  One 16x16 tile per wave.
// grid.x = N/16, block = 128 (4 waves -> 4 M-tiles of the fixed M=64)
// ---------------------------------------------------------------------------
__device__ __forceinline__ v16bf load_a_frag(const float* __restrict__ A, int lda,
                                             int mbase, int k0, int lane) {
  // 16-bit A 16x32 layout: lane = half*16 + m; VGPR v: K = 16*(v>=4) + 2*(v&3) + 8*half
  const int m    = mbase + (lane & 15);
  const int half = lane >> 4;
  const float* base = A + (size_t)m * lda + k0 + half * 8;
  v16bf a;
#pragma unroll
  for (int v = 0; v < 8; ++v) {
    const int kb = ((v & 3) << 1) + ((v >> 2) << 4);
    float2 f2 = *(const float2*)(base + kb);
    a[2 * v]     = (__bf16)f2.x;
    a[2 * v + 1] = (__bf16)f2.y;
  }
  return a;
}

__device__ __forceinline__ v16bf load_b_frag(const float* __restrict__ W, int ldw,
                                             int nbase, int k0, int lane) {
  // B 32x16 layout: lane = half*16 + n; VGPR v: K = 16*(v>=4) + 2*(v&3) + 8*half
  const int n    = nbase + (lane & 15);
  const int half = lane >> 4;
  const float* base = W + (size_t)n * ldw + k0 + half * 8;
  v16bf bf;
#pragma unroll
  for (int v = 0; v < 8; ++v) {
    float2 f2 = *(const float2*)(base + ((v & 3) << 1) + ((v >> 2) << 4));
    bf[2 * v]     = (__bf16)f2.x;
    bf[2 * v + 1] = (__bf16)f2.y;
  }
  return bf;
}

__global__ void __launch_bounds__(128)
wmma_gemm_kernel(const float* __restrict__ A0, int lda0, int K0,
                 const float* __restrict__ A1, int lda1, int K1,
                 const float* __restrict__ W0,
                 const float* __restrict__ W1,
                 const float* __restrict__ bias,
                 float* __restrict__ Out, int ldo)
{
  const int lane  = threadIdx.x & 31;
  const int wave  = threadIdx.x >> 5;
  const int nbase = blockIdx.x * 16;
  const int mbase = wave * 16;
  const int col   = lane & 15;
  const int half  = lane >> 4;

  v8f acc;
  const float bv = bias ? bias[nbase + col] : 0.f;
#pragma unroll
  for (int r = 0; r < 8; ++r) acc[r] = bv;

  for (int k0 = 0; k0 < K0; k0 += 32) {
    v16bf a = load_a_frag(A0, lda0, mbase, k0, lane);
    v16bf b = load_b_frag(W0, K0, nbase, k0, lane);
    acc = __builtin_amdgcn_wmma_f32_16x16x32_bf16(false, a, false, b,
                                                  (short)0, acc, false, false);
  }
  if (A1) {
    for (int k0 = 0; k0 < K1; k0 += 32) {
      v16bf a = load_a_frag(A1, lda1, mbase, k0, lane);
      v16bf b = load_b_frag(W1, K1, nbase, k0, lane);
      acc = __builtin_amdgcn_wmma_f32_16x16x32_bf16(false, a, false, b,
                                                    (short)0, acc, false, false);
    }
  }
  // C/D layout: VGPR r -> row (r + 8*half), col = lane&15
  float* orow = Out + nbase + col;
#pragma unroll
  for (int r = 0; r < 8; ++r)
    orow[(size_t)(mbase + half * 8 + r) * ldo] = acc[r];
}

// ---------------------------------------------------------------------------
// LSTM gate fusion: g(B,4H) -> h (and optional c).  h = o*tanh(f*c_in + i*tanh(g))
// c_in == nullptr means layers 2..4 (h_prev = c_prev = 0)
// ---------------------------------------------------------------------------
__global__ void __launch_bounds__(256)
lstm_gate_kernel(const float* __restrict__ g,
                 const float* __restrict__ b_ih,
                 const float* __restrict__ b_hh,
                 const float* __restrict__ c_in,
                 float* __restrict__ h_out,
                 float* __restrict__ h_out2,
                 float* __restrict__ c_out)
{
  const int idx = blockIdx.x * blockDim.x + threadIdx.x;   // b*H + j
  if (idx >= BATCH * HDIM) return;
  const int b = idx >> 9;
  const int j = idx & (HDIM - 1);
  const float* gr = g + (size_t)b * 4 * HDIM;

  const float gi = gr[j]            + b_ih[j]            + b_hh[j];
  const float gf = gr[HDIM + j]     + b_ih[HDIM + j]     + b_hh[HDIM + j];
  const float gg = gr[2 * HDIM + j] + b_ih[2 * HDIM + j] + b_hh[2 * HDIM + j];
  const float go = gr[3 * HDIM + j] + b_ih[3 * HDIM + j] + b_hh[3 * HDIM + j];

  const float i = 1.f / (1.f + __expf(-gi));
  const float f = 1.f / (1.f + __expf(-gf));
  const float o = 1.f / (1.f + __expf(-go));
  const float t = tanhf(gg);
  const float c = c_in ? fmaf(f, c_in[idx], i * t) : (i * t);
  const float h = o * tanhf(c);

  h_out[idx] = h;
  if (h_out2) h_out2[idx] = h;
  if (c_out)  c_out[idx]  = c;
}

// ---------------------------------------------------------------------------
extern "C" void kernel_launch(void* const* d_in, const int* in_sizes, int n_in,
                              void* d_out, int out_size, void* d_ws, size_t ws_size,
                              hipStream_t stream) {
  const int*   x     = (const int*)  d_in[0];
  const float* hsd   = (const float*)d_in[1];
  const float* hse   = (const float*)d_in[2];
  const float* h0    = (const float*)d_in[3];   // (1,B,H)
  const float* c0    = (const float*)d_in[4];   // (1,B,H)
  const float* emb   = (const float*)d_in[5];
  const float* w_ih1 = (const float*)d_in[6];   // (4H, E+H)
  const float* w_hh1 = (const float*)d_in[7];   // (4H, H)
  const float* b_ih1 = (const float*)d_in[8];
  const float* b_hh1 = (const float*)d_in[9];
  const float* w_ih  = (const float*)d_in[10];  // (3, 4H, H)
  const float* b_ih  = (const float*)d_in[12];  // (3, 4H)
  const float* b_hh  = (const float*)d_in[13];
  const float* fc_w  = (const float*)d_in[14];  // (V, H)
  const float* fc_b  = (const float*)d_in[15];
  (void)in_sizes; (void)n_in; (void)out_size; (void)ws_size; (void)d_in;

  float* out    = (float*)d_out;
  float* logits = out;                         // B * V
  float* h_out  = out + (size_t)BATCH * VOC;   // B * H
  float* c_out  = h_out + BATCH * HDIM;        // B * H

  float* ws   = (float*)d_ws;
  float* inp  = ws;                              // B * (E+H)     = 49152
  float* g    = inp  + BATCH * (EDIM + HDIM);    // B * 4H        = 131072
  float* z    = g    + BATCH * 4 * HDIM;         // B * H         = 32768
  float* pm   = z    + BATCH * HDIM;             // B * NSPLIT
  float* pl   = pm   + BATCH * NSPLIT;
  float* pacc = pl   + BATCH * NSPLIT;           // B*NSPLIT*H    = 262144

  // 1) attention (single non-temporal HBM pass over 256 MB) + embedding
  attn_partial_kernel<<<BATCH * NSPLIT, 256, 0, stream>>>(hsd, hse, pm, pl, pacc);
  attn_finalize_kernel<<<BATCH, 256, 0, stream>>>(x, emb, pm, pl, pacc, inp);

  // 2) LSTM layer 1: g = inp @ w_ih1^T + h0 @ w_hh1^T ; gates with real c0
  wmma_gemm_kernel<<<4 * HDIM / 16, 128, 0, stream>>>(
      inp, EDIM + HDIM, EDIM + HDIM, h0, HDIM, HDIM, w_ih1, w_hh1, nullptr,
      g, 4 * HDIM);
  lstm_gate_kernel<<<(BATCH * HDIM) / 256, 256, 0, stream>>>(
      g, b_ih1, b_hh1, c0, z, nullptr, nullptr);

  // 3) LSTM layers 2..4: h_prev = c_prev = 0 => skip w_hh term entirely
  for (int L = 0; L < 3; ++L) {
    wmma_gemm_kernel<<<4 * HDIM / 16, 128, 0, stream>>>(
        z, HDIM, HDIM, nullptr, 0, 0, w_ih + (size_t)L * 4 * HDIM * HDIM,
        nullptr, nullptr, g, 4 * HDIM);
    const bool last = (L == 2);
    lstm_gate_kernel<<<(BATCH * HDIM) / 256, 256, 0, stream>>>(
        g, b_ih + L * 4 * HDIM, b_hh + L * 4 * HDIM, nullptr,
        z, last ? h_out : nullptr, last ? c_out : nullptr);
  }

  // 4) logits = z @ fc_w^T + fc_b   (M=64, N=32000, K=512)
  wmma_gemm_kernel<<<VOC / 16, 128, 0, stream>>>(
      z, HDIM, HDIM, nullptr, 0, 0, fc_w, nullptr, fc_b, logits, VOC);
}
